// LiquidObserver_45260365365707
// MI455X (gfx1250) — matrix operand order repaired and checked by reading, not compiled, not run on hardware
//
#include <hip/hip_runtime.h>

typedef __bf16 bf16;
typedef __attribute__((ext_vector_type(16))) __bf16 v16bf;
typedef __attribute__((ext_vector_type(8)))  __bf16 v8bf;
typedef __attribute__((ext_vector_type(8)))  float  v8f;

#define BB   32
#define SS   512
#define DIN  1024
#define HH   512
#define OUTN 1024
#define LDW  520   // LDS row stride (bf16): 1040B = 16B-aligned, rotates banks

#if __has_builtin(__builtin_amdgcn_sched_group_barrier)
#define SCHED_GROUP(mask, size, id) __builtin_amdgcn_sched_group_barrier(mask, size, id)
#else
#define SCHED_GROUP(mask, size, id)
#endif

// ---------------------------------------------------------------------------
// CDNA5 wave32 WMMA fragment layouts used here (A/B use contiguous K, not the ISA table order):
//  A (16x32, MxK):  lane L row M=L&15; K = {kb..kb+15},
//                   kb = k0 + ((L&16)?16:0)     -> two 16B loads
//  B (32x16, KxN):  lane L col N=L&15; 16 contiguous K at k0+((L&16)?16:0)
//                   -> one 32B load from the pre-transposed weight [N,K]
//  C/D (16x16 f32): lane L col N=L&15; VGPR v row M = v + ((L&16)?8:0)
// ---------------------------------------------------------------------------

__device__ __forceinline__ v8f wmma_bf16(v16bf a, v16bf b, v8f c) {
  return __builtin_amdgcn_wmma_f32_16x16x32_bf16(false, a, false, b,
                                                 (short)0, c, false, false);
}

// A-fragment from a per-lane row base (base already includes m*ld + akb)
__device__ __forceinline__ v16bf ldsA(const bf16* __restrict__ row, int k0) {
  v8bf lo = *(const v8bf*)(row + k0);
  v8bf hi = *(const v8bf*)(row + k0 + 8);
  v16bf f;
#pragma unroll
  for (int i = 0; i < 8; ++i) { f[i] = lo[i]; f[i + 8] = hi[i]; }
  return f;
}

// A-fragment from bf16 global, row-major [M,lda]
__device__ __forceinline__ v16bf load_afrag(const bf16* __restrict__ A, int lda,
                                            int m0, int k0, int lane) {
  int m  = m0 + (lane & 15);
  int kb = k0 + ((lane & 16) ? 16 : 0);
  const bf16* p = A + (size_t)m * lda + kb;
  v8bf lo = *(const v8bf*)p;
  v8bf hi = *(const v8bf*)(p + 8);
  v16bf f;
#pragma unroll
  for (int i = 0; i < 8; ++i) { f[i] = lo[i]; f[i + 8] = hi[i]; }
  return f;
}

// A-fragment from f32 global with on-the-fly bf16 conversion
__device__ __forceinline__ v16bf load_afrag(const float* __restrict__ A, int lda,
                                            int m0, int k0, int lane) {
  int m  = m0 + (lane & 15);
  int kb = k0 + ((lane & 16) ? 16 : 0);
  const float* p = A + (size_t)m * lda + kb;
  v16bf f;
#pragma unroll
  for (int i = 0; i < 8; ++i) { f[i] = (bf16)p[i]; f[i + 8] = (bf16)p[i + 8]; }
  return f;
}

// ---- branch-free activations (v_exp_f32 / v_rcp_f32, no saveexec) ---------
__device__ __forceinline__ float fast_sigmoid(float x) {
  return __builtin_amdgcn_rcpf(1.0f + __expf(-x));
}
__device__ __forceinline__ float fast_tanh(float x) {
  float e = __expf(-2.0f * fabsf(x));          // in (0,1]: overflow-free
  float t = (1.0f - e) * __builtin_amdgcn_rcpf(1.0f + e);
  return copysignf(t, x);
}
__device__ __forceinline__ float gelu_tanh(float x) {
  float x3 = x * x * x;
  return 0.5f * x * (1.0f + fast_tanh(0.7978845608028654f * (x + 0.044715f * x3)));
}

// ---------------------------------------------------------------------------
// Weight convert + transpose: src [K,N] f32 -> dst [N,Kpad] bf16 (zero-pad K)
// ---------------------------------------------------------------------------
__global__ void lo_cvt_t(const float* __restrict__ src, bf16* __restrict__ dst,
                         int K, int N, int Kpad) {
  int i = blockIdx.x * blockDim.x + threadIdx.x;
  if (i >= N * Kpad) return;
  int n = i / Kpad, k = i % Kpad;
  dst[i] = (k < K) ? (bf16)src[(size_t)k * N + n] : (bf16)0.0f;
}

// ---------------------------------------------------------------------------
// WMMA GEMM, one wave per 16x16 tile, 2-deep pipelined K loop (K%32==0).
// MODE 0: +bias->bf16   MODE 1: +bias,GELU->bf16   MODE 2: +bias->f32
// ---------------------------------------------------------------------------
template <int MODE, typename AT>
__global__ __launch_bounds__(256) void lo_gemm(const AT* __restrict__ A,
                                               const bf16* __restrict__ BT,
                                               const float* __restrict__ bias,
                                               void* __restrict__ out,
                                               int M, int N, int K) {
  int lane = threadIdx.x & 31;
  int wid  = threadIdx.x >> 5;
  int tile = blockIdx.x * 8 + wid;
  int nt_n = N >> 4;
  int mt = tile / nt_n, nt = tile % nt_n;
  if (mt >= (M >> 4)) return;
  int m0 = mt << 4, n0 = nt << 4;

  const bf16* bbase = BT + (size_t)(n0 + (lane & 15)) * K + ((lane & 16) ? 16 : 0);

  v8f c = {};
  v16bf a = load_afrag(A, K, m0, 0, lane);
  v16bf b = *(const v16bf*)bbase;
#pragma unroll 2
  for (int k0 = 32; k0 < K; k0 += 32) {
    v16bf an = load_afrag(A, K, m0, k0, lane);
    v16bf bn = *(const v16bf*)(bbase + k0);
    c = wmma_bf16(a, b, c);
    a = an; b = bn;
  }
  c = wmma_bf16(a, b, c);

  int ncol  = n0 + (lane & 15);
  float bv  = bias[ncol];
  int mbase = m0 + ((lane & 16) ? 8 : 0);
#pragma unroll
  for (int v = 0; v < 8; ++v) {
    float x = c[v] + bv;
    size_t off = (size_t)(mbase + v) * N + ncol;
    if (MODE == 0)      ((bf16*)out)[off]  = (bf16)x;
    else if (MODE == 1) ((bf16*)out)[off]  = (bf16)gelu_tanh(x);
    else                ((float*)out)[off] = x;
  }
}

// ---------------------------------------------------------------------------
// Persistent CfC scan. 2 blocks (batch tiles of 16) x 1024 threads (32 waves).
// Double-buffered x and h in LDS -> ONE barrier per step.  Wave w owns hidden
// columns [16w,16w+16): 4 WMMA per K-tile x 32 K-tiles, fully unrolled.
// Gate weights stream from L2 every step (laundered zero offset defeats LICM
// while keeping global addrspace -> global_load_b128, LOADcnt only).
// sched_group_barrier pins each k-tile to: 8 VMEM-reads (next tile's weights)
// -> 2 DS-reads (A frag) -> 4 WMMA (current tile), so a full k-tile of weight
// loads stays in flight under the matrix ops instead of being sunk to
// just-in-time loads with s_wait_loadcnt 0x0 before every WMMA.
// ---------------------------------------------------------------------------
__global__ __launch_bounds__(1024) void lo_cfc(
    const bf16* __restrict__ proj,                      // [B*S, H] bf16
    const bf16* __restrict__ Wff1T, const bf16* __restrict__ Wff2T,
    const bf16* __restrict__ WtaT,  const bf16* __restrict__ WtbT, // [H, 2H]
    const float* __restrict__ bff1, const float* __restrict__ bff2,
    const float* __restrict__ bta,  const float* __restrict__ btb,
    bf16* __restrict__ cfc)                             // [B*S, H] bf16
{
  __shared__ bf16 xsh[2][16][LDW];   // x_t double buffer
  __shared__ bf16 hsh[2][16][LDW];   // h   double buffer

  const int tid  = threadIdx.x;
  const int lane = tid & 31;
  const int wid  = tid >> 5;
  const int m0   = blockIdx.x * 16;     // batch base
  const int n0   = wid * 16;            // this wave's hidden-column slice
  const int xr   = (tid * 8) >> 9;      // staging row   (16 rows x 512 cols)
  const int xc8  = (tid * 8) & 511;     // staging col

  // prologue: stage x_0, zero h_0
  *(v8bf*)&xsh[0][xr][xc8] =
      *(const v8bf*)(proj + ((size_t)(m0 + xr) * SS + 0) * HH + xc8);
  { v8bf z = {}; *(v8bf*)&hsh[0][xr][xc8] = z; }

  // loop-invariant per-lane values
  const int   nc   = n0 + (lane & 15);
  const float bf1  = bff1[nc], bf2 = bff2[nc], btab = bta[nc] + btb[nc];
  const int   mb   = (lane & 16) ? 8 : 0;
  const int   am   = lane & 15;                  // A-fragment row
  const int   akb  = (lane & 16) ? 16 : 0;       // A-fragment K sub-offset
  const size_t boff = (size_t)nc * (2 * HH) + ((lane & 16) ? 16 : 0);
  const bf16* g1 = Wff1T + boff;                 // per-lane B bases
  const bf16* g2 = Wff2T + boff;
  const bf16* g3 = WtaT  + boff;
  const bf16* g4 = WtbT  + boff;
  __syncthreads();

  for (int t = 0; t < SS; ++t) {
    const int cur = t & 1, nxt = cur ^ 1;
    const bf16* xrow = &xsh[cur][am][akb];
    const bf16* hrow = &hsh[cur][am][akb];

    // Laundered zero offset: opaque to LICM (loads stay inside the t-loop,
    // re-streamed from L2) but preserves global addrspace of g1..g4.
    int zoff = 0;
    asm volatile("" : "+v"(zoff));
    const bf16* w1 = g1 + zoff;
    const bf16* w2 = g2 + zoff;
    const bf16* w3 = g3 + zoff;
    const bf16* w4 = g4 + zoff;

    // issue next timestep's x load early; latency hides under the K loop
    int tn = (t + 1 < SS) ? (t + 1) : t;
    v8bf xreg = *(const v8bf*)(proj + ((size_t)(m0 + xr) * SS + tn) * HH + xc8);

    v8f ff1 = {}, ff2 = {}, ta = {}, tb = {};

    // fully unrolled K loop (k 0..511 = x, 512..1023 = h); global B-fragments
    // pipelined 2-deep; schedule pinned per k-tile.
    v16bf b1 = *(const v16bf*)w1;
    v16bf b2 = *(const v16bf*)w2;
    v16bf b3 = *(const v16bf*)w3;
    v16bf b4 = *(const v16bf*)w4;
#pragma unroll
    for (int kt = 0; kt < 31; ++kt) {
      const int k0  = kt * 32;
      const int k0n = k0 + 32;
      v16bf a  = (kt < 16) ? ldsA(xrow, k0) : ldsA(hrow, k0 - 512);
      v16bf c1 = *(const v16bf*)(w1 + k0n);
      v16bf c2 = *(const v16bf*)(w2 + k0n);
      v16bf c3 = *(const v16bf*)(w3 + k0n);
      v16bf c4 = *(const v16bf*)(w4 + k0n);
      ff1 = wmma_bf16(a, b1, ff1);
      ff2 = wmma_bf16(a, b2, ff2);
      ta  = wmma_bf16(a, b3, ta);
      tb  = wmma_bf16(a, b4, tb);
      b1 = c1; b2 = c2; b3 = c3; b4 = c4;
      // pin schedule: next-tile weight loads first, then A frag, then WMMAs
      SCHED_GROUP(0x020, 8, 0);   // 8 x VMEM read  (c1..c4)
      SCHED_GROUP(0x100, 2, 0);   // 2 x DS read    (a)
      SCHED_GROUP(0x008, 4, 0);   // 4 x WMMA
    }
    {
      v16bf a = ldsA(hrow, 31 * 32 - 512);
      ff1 = wmma_bf16(a, b1, ff1);
      ff2 = wmma_bf16(a, b2, ff2);
      ta  = wmma_bf16(a, b3, ta);
      tb  = wmma_bf16(a, b4, tb);
    }

    // stage x_{t+1} into the write buffer
    *(v8bf*)&xsh[nxt][xr][xc8] = xreg;

    // fused CfC update: h = tanh(ff1) + sigma(ta+tb)*(tanh(ff2)-tanh(ff1))
#pragma unroll
    for (int v = 0; v < 8; ++v) {
      float s  = fast_sigmoid(ta[v] + tb[v] + btab);
      float f1 = fast_tanh(ff1[v] + bf1);
      float f2 = fast_tanh(ff2[v] + bf2);
      bf16 hb = (bf16)(f1 + s * (f2 - f1));
      int mr = mb + v;
      hsh[nxt][mr][nc] = hb;                                // state feedback
      cfc[((size_t)(m0 + mr) * SS + t) * HH + nc] = hb;     // motor output
    }
    __syncthreads();   // flip buffers: nxt writes -> visible as next cur
  }
}

// ---------------------------------------------------------------------------
extern "C" void kernel_launch(void* const* d_in, const int* in_sizes, int n_in,
                              void* d_out, int out_size, void* d_ws, size_t ws_size,
                              hipStream_t stream) {
  const float* X    = (const float*)d_in[0];
  const float* Wp   = (const float*)d_in[1];
  const float* bp   = (const float*)d_in[2];
  const float* Wff1 = (const float*)d_in[3];
  const float* bff1 = (const float*)d_in[4];
  const float* Wff2 = (const float*)d_in[5];
  const float* bff2 = (const float*)d_in[6];
  const float* Wta  = (const float*)d_in[7];
  const float* bta  = (const float*)d_in[8];
  const float* Wtb  = (const float*)d_in[9];
  const float* btb  = (const float*)d_in[10];
  const float* Wh1  = (const float*)d_in[11];
  const float* bh1  = (const float*)d_in[12];
  const float* Wh2  = (const float*)d_in[13];
  const float* bh2  = (const float*)d_in[14];
  float* out = (float*)d_out;

  // workspace carve-up (bf16, 256B aligned)
  char* ws = (char*)d_ws;
  size_t off = 0;
  auto carve = [&](size_t elems) {
    void* p = ws + off;
    off = (off + elems * sizeof(bf16) + 255) & ~(size_t)255;
    return (bf16*)p;
  };
  bf16* WpT   = carve((size_t)HH * DIN);        // [512,1024]
  bf16* Wff1T = carve((size_t)HH * 2 * HH);     // [512,1024]
  bf16* Wff2T = carve((size_t)HH * 2 * HH);
  bf16* WtaT  = carve((size_t)HH * 2 * HH);
  bf16* WtbT  = carve((size_t)HH * 2 * HH);
  bf16* Wh1T  = carve((size_t)HH * HH);         // [512,512], K 509 zero-padded
  bf16* Wh2T  = carve((size_t)OUTN * HH);       // [1024,512]
  bf16* projb = carve((size_t)BB * SS * HH);    // [16384,512]
  bf16* cfcb  = carve((size_t)BB * SS * HH);    // [16384,512]
  bf16* hmid  = projb;                          // reuse: proj dead after scan

  // 1) convert + transpose weights to bf16 B-layout
  auto cvt = [&](const float* s, bf16* d, int K, int N, int Kpad) {
    int total = N * Kpad;
    lo_cvt_t<<<(total + 255) / 256, 256, 0, stream>>>(s, d, K, N, Kpad);
  };
  cvt(Wp,   WpT,   DIN,    HH,   DIN);
  cvt(Wff1, Wff1T, 2 * HH, HH,   2 * HH);
  cvt(Wff2, Wff2T, 2 * HH, HH,   2 * HH);
  cvt(Wta,  WtaT,  2 * HH, HH,   2 * HH);
  cvt(Wtb,  WtbT,  2 * HH, HH,   2 * HH);
  cvt(Wh1,  Wh1T,  509,    HH,   HH);
  cvt(Wh2,  Wh2T,  HH,     OUTN, HH);

  const int M = BB * SS;  // 16384

  // 2) input projection
  lo_gemm<0, float><<<(M / 16) * (HH / 16) / 8, 256, 0, stream>>>(
      X, WpT, bp, projb, M, HH, DIN);

  // 3) persistent CfC scan
  lo_cfc<<<2, 1024, 0, stream>>>(projb, Wff1T, Wff2T, WtaT, WtbT,
                                 bff1, bff2, bta, btb, cfcb);

  // 4) head: hmid = gelu(cfc @ Wh1 + bh1)
  lo_gemm<1, bf16><<<(M / 16) * (HH / 16) / 8, 256, 0, stream>>>(
      cfcb, Wh1T, bh1, hmid, M, HH, HH);

  // 5) head: out = hmid @ Wh2 + bh2 -> f32
  lo_gemm<2, bf16><<<(M / 16) * (OUTN / 16) / 8, 256, 0, stream>>>(
      hmid, Wh2T, bh2, out, M, OUTN, HH);
}
